// ASM_5196910428634
// MI455X (gfx1250) — matrix operand order repaired and checked by reading, not compiled, not run on hardware
//
#include <hip/hip_runtime.h>

// ---------------------------------------------------------------------------
// Non-local attention block for MI455X (gfx1250, wave32, WMMA).
// All GEMMs use v_wmma_f32_16x16x32_f16; attention is flash-style fused
// (scores never hit memory). Intermediates are f16 in workspace.
// ---------------------------------------------------------------------------

typedef __attribute__((ext_vector_type(16))) _Float16 v16h;
typedef __attribute__((ext_vector_type(8)))  float    v8f;

#define NB   8
#define CIN  256
#define CI   128
#define NSP  4096   // 64*64
#define NK   1024   // 32*32

#define WMMA_F16(A, B, C) \
  __builtin_amdgcn_wmma_f32_16x16x32_f16(false, (A), false, (B), (short)0, (C), false, false)

// ---------------------------------------------------------------------------
// Projection: out[o][q] = sum_c w[o][c] * fuse[b][c][q] + bias[o]  (f16 out)
// One wave per (batch, projection, 16-wide q tile); wave computes all 8
// 16-row output tiles (CI=128) so the fuse B-tile is loaded once per k-chunk.
// Output layout: [B][CI][NSP] (spatial contiguous).
// ---------------------------------------------------------------------------
__global__ void __launch_bounds__(256) proj_kernel(
    const float* __restrict__ fuse,
    const float* __restrict__ th_w, const float* __restrict__ th_b,
    const float* __restrict__ g_w,  const float* __restrict__ g_b,
    const float* __restrict__ ph_w, const float* __restrict__ ph_b,
    _Float16* __restrict__ theta_h, _Float16* __restrict__ g_full,
    _Float16* __restrict__ phi_full)
{
  const int wave = (blockIdx.x * blockDim.x + threadIdx.x) >> 5;
  const int lane = threadIdx.x & 31;
  const int lm   = lane & 15;
  const bool hi  = lane >= 16;

  const int qt   = wave & 255;          // q tile
  const int bp   = wave >> 8;
  const int proj = bp % 3;
  const int b    = bp / 3;
  const int qbase = qt * 16;

  const float* w; const float* bias; _Float16* outp;
  if (proj == 0)      { w = th_w; bias = th_b; outp = theta_h; }
  else if (proj == 1) { w = g_w;  bias = g_b;  outp = g_full; }
  else                { w = ph_w; bias = ph_b; outp = phi_full; }

  v8f acc[8] = {};
  const float* xsrc = fuse + (size_t)b * CIN * NSP + qbase;

  for (int kc = 0; kc < 8; ++kc) {            // K = 256 channels, 8 chunks of 32
    // B operand: lane holds column n = q = lm; e0-7 = K{0-7|8-15}, e8-15 = K{16-23|24-31}
    const float* bp0 = xsrc + (size_t)(kc * 32 + (hi ? 8 : 0)) * NSP + lm;
    v16h Bv;
    #pragma unroll
    for (int e = 0; e < 8; ++e) { Bv[e] = (_Float16)bp0[(size_t)e * NSP]; Bv[8 + e] = (_Float16)bp0[(size_t)(16 + e) * NSP]; }
    #pragma unroll
    for (int mt = 0; mt < 8; ++mt) {
      // A operand: lane holds row m = o; e0-7 = K{0-7|8-15}, e8-15 = K{16-23|24-31}
      const int o = mt * 16 + lm;
      const float* ar = w + (size_t)o * CIN + kc * 32 + (hi ? 8 : 0);
      v16h Av;
      #pragma unroll
      for (int e = 0; e < 8; ++e) { Av[e] = (_Float16)ar[e]; Av[8 + e] = (_Float16)ar[16 + e]; }
      acc[mt] = WMMA_F16(Av, Bv, acc[mt]);
    }
  }
  // epilogue: bias, convert, store [CI][NSP]
  const int q = qbase + lm;
  #pragma unroll
  for (int mt = 0; mt < 8; ++mt) {
    #pragma unroll
    for (int r = 0; r < 8; ++r) {
      const int o = mt * 16 + r + (hi ? 8 : 0);
      outp[(size_t)b * CI * NSP + (size_t)o * NSP + q] = (_Float16)(acc[mt][r] + bias[o]);
    }
  }
}

// ---------------------------------------------------------------------------
// 2x2 max pool + transpose: [B][CI][64][64] f16 -> [B][NK][CI] f16
// ---------------------------------------------------------------------------
__global__ void pool_kernel(const _Float16* __restrict__ full,
                            _Float16* __restrict__ pooled)
{
  const int idx = blockIdx.x * blockDim.x + threadIdx.x;   // NB*CI*NK threads
  if (idx >= NB * CI * NK) return;
  const int c    = idx & (CI - 1);
  const int rest = idx >> 7;
  const int key  = rest & (NK - 1);
  const int b    = rest >> 10;
  const int ky = key >> 5, kx = key & 31;
  const _Float16* src = full + ((size_t)b * CI + c) * NSP + (ky * 2) * 64 + kx * 2;
  const float v0 = (float)src[0],  v1 = (float)src[1];
  const float v2 = (float)src[64], v3 = (float)src[65];
  pooled[((size_t)b * NK + key) * CI + c] =
      (_Float16)fmaxf(fmaxf(v0, v1), fmaxf(v2, v3));
}

// ---------------------------------------------------------------------------
// Flash attention: one wave per (batch, 16-query tile).
//   S^T[key][q] = sum_c phi[key][c] * theta[c][q]   (A = phi rows, B = theta)
//   online softmax over keys; two adjacent 16-key score tiles pack lane-local
//   into the A operand of the 16x16x32 PV WMMA (no cross-lane transpose).
//   y[q][c] += P[q][key] * g[key][c]; output y as [B][NSP][CI] f16.
// ---------------------------------------------------------------------------
__global__ void __launch_bounds__(256) attn_kernel(
    const _Float16* __restrict__ theta_h,  // [B][CI][NSP]
    const _Float16* __restrict__ phi_h,    // [B][NK][CI]
    const _Float16* __restrict__ g_h,      // [B][NK][CI]
    _Float16* __restrict__ y_h)            // [B][NSP][CI]
{
  const int wave = (blockIdx.x * blockDim.x + threadIdx.x) >> 5;
  const int lane = threadIdx.x & 31;
  const int lm   = lane & 15;
  const bool hi  = lane >= 16;
  const int qt   = wave & 255;
  const int b    = wave >> 8;
  const int qbase = qt * 16;

  // Preload B operand of the score GEMM: theta chunks (K = 128 channels).
  v16h Bth[4];
  {
    const _Float16* tb = theta_h + (size_t)b * CI * NSP + qbase;
    #pragma unroll
    for (int kc = 0; kc < 4; ++kc) {
      const _Float16* p = tb + (size_t)(kc * 32 + (hi ? 8 : 0)) * NSP + lm; // lane = q column
      #pragma unroll
      for (int e = 0; e < 8; ++e) { Bth[kc][e] = p[(size_t)e * NSP]; Bth[kc][8 + e] = p[(size_t)(16 + e) * NSP]; }
    }
  }

  v8f yacc[8] = {};
  float m_run = -1e30f, l_run = 0.f;
  const _Float16* phib = phi_h + (size_t)b * NK * CI;
  const _Float16* gb   = g_h   + (size_t)b * NK * CI;

  for (int k0 = 0; k0 < NK; k0 += 32) {
    // --- two 16-key S^T tiles ---
    v8f st[2];
    #pragma unroll
    for (int t = 0; t < 2; ++t) {
      v8f s = {};
      #pragma unroll
      for (int kc = 0; kc < 4; ++kc) {
        const _Float16* ar = phib + (size_t)(k0 + t * 16 + lm) * CI
                                  + kc * 32 + (hi ? 8 : 0);
        v16h Av;
        #pragma unroll
        for (int e = 0; e < 8; ++e) { Av[e] = ar[e]; Av[8 + e] = ar[16 + e]; }
        s = WMMA_F16(Av, Bth[kc], s);
      }
      st[t] = s;
    }
    // --- online softmax (per query q = lane%16; pairs of lanes l, l^16) ---
    float mloc = -1e30f;
    #pragma unroll
    for (int r = 0; r < 8; ++r) mloc = fmaxf(mloc, fmaxf(st[0][r], st[1][r]));
    mloc = fmaxf(mloc, __shfl_xor(mloc, 16));
    const float m_new   = fmaxf(m_run, mloc);
    const float rescale = __expf(m_run - m_new);
    float psum = 0.f;
    v16h Pa;                                  // A operand of PV WMMA (K=32 keys)
    #pragma unroll
    for (int r = 0; r < 8; ++r) {
      const float p0 = __expf(st[0][r] - m_new);
      const float p1 = __expf(st[1][r] - m_new);
      psum += p0 + p1;
      Pa[r]     = (_Float16)p0;               // keys k0+{0-7|8-15}
      Pa[8 + r] = (_Float16)p1;               // keys k0+16+{0-7|8-15}
    }
    psum += __shfl_xor(psum, 16);
    l_run = l_run * rescale + psum;
    m_run = m_new;
    // rescale y accumulators: element r belongs to query q = r (+8 for hi lanes)
    float fs[8];
    #pragma unroll
    for (int r = 0; r < 8; ++r) fs[r] = __shfl(rescale, r + (hi ? 8 : 0));
    #pragma unroll
    for (int ct = 0; ct < 8; ++ct)
      #pragma unroll
      for (int r = 0; r < 8; ++r) yacc[ct][r] *= fs[r];
    // --- PV: y[16q x 16c] += P[16q x 32key] * g[32key x 16c] ---
    #pragma unroll
    for (int ct = 0; ct < 8; ++ct) {
      const _Float16* gr = gb + (size_t)(k0 + (hi ? 8 : 0)) * CI + ct * 16 + lm; // lane = c column
      v16h Gv;
      #pragma unroll
      for (int e = 0; e < 8; ++e) { Gv[e] = gr[(size_t)e * CI]; Gv[8 + e] = gr[(size_t)(16 + e) * CI]; }
      yacc[ct] = WMMA_F16(Pa, Gv, yacc[ct]);
    }
  }
  // finalize: divide by softmax denominator, store y as [NSP][CI]
  const float invl = 1.f / l_run;
  float fs[8];
  #pragma unroll
  for (int r = 0; r < 8; ++r) fs[r] = __shfl(invl, r + (hi ? 8 : 0));
  _Float16* yo = y_h + (size_t)b * NSP * CI;
  #pragma unroll
  for (int ct = 0; ct < 8; ++ct)
    #pragma unroll
    for (int r = 0; r < 8; ++r) {
      const int q = qbase + r + (hi ? 8 : 0);
      yo[(size_t)q * CI + ct * 16 + lm] = (_Float16)(yacc[ct][r] * fs[r]);
    }
}

// ---------------------------------------------------------------------------
// Final 1x1 conv + BN + residual, written into the middle channel block of
// the concatenated output. One wave per (batch, half of C, 16-q tile).
// z[o][q] = BN( sum_c W_w[o][c]*y[q][c] + W_b[o] ) + fuse[b][o][q]
// ---------------------------------------------------------------------------
__global__ void __launch_bounds__(256) wout_kernel(
    const _Float16* __restrict__ y_h,     // [B][NSP][CI]
    const float* __restrict__ W_w,        // [C][CI]
    const float* __restrict__ W_b,
    const float* __restrict__ bn_gamma, const float* __restrict__ bn_beta,
    const float* __restrict__ bn_mean,  const float* __restrict__ bn_var,
    const float* __restrict__ fuse,
    float* __restrict__ out)              // [B][3C][NSP]
{
  const int wave = (blockIdx.x * blockDim.x + threadIdx.x) >> 5;
  const int lane = threadIdx.x & 31;
  const int lm   = lane & 15;
  const bool hi  = lane >= 16;
  const int qt   = wave & 255;
  const int rest = wave >> 8;
  const int oh   = rest & 1;
  const int b    = rest >> 1;
  const int qbase = qt * 16;

  // Preload B operand: y chunks for this q tile (K = 128 channels)
  v16h By[4];
  const _Float16* yb = y_h + (size_t)b * NSP * CI;
  #pragma unroll
  for (int kc = 0; kc < 4; ++kc) {
    const int c = kc * 32 + (hi ? 8 : 0);            // lane = q column
    #pragma unroll
    for (int e = 0; e < 8; ++e) { By[kc][e] = yb[(size_t)(qbase + lm) * CI + c + e]; By[kc][8 + e] = yb[(size_t)(qbase + lm) * CI + c + 16 + e]; }
  }

  v8f acc[8] = {};
  #pragma unroll
  for (int kc = 0; kc < 4; ++kc) {
    #pragma unroll
    for (int mt = 0; mt < 8; ++mt) {
      const int o = oh * 128 + mt * 16 + lm;
      const float* ar = W_w + (size_t)o * CI + kc * 32 + (hi ? 8 : 0);
      v16h Av;
      #pragma unroll
      for (int e = 0; e < 8; ++e) { Av[e] = (_Float16)ar[e]; Av[8 + e] = (_Float16)ar[16 + e]; }
      acc[mt] = WMMA_F16(Av, By[kc], acc[mt]);
    }
  }
  // epilogue: bias + BN(eval) + residual, write z into out channels [256,512)
  const int q = qbase + lm;
  const float* xb = fuse + (size_t)b * CIN * NSP;
  float* ob = out + ((size_t)b * 3 * CIN + CIN) * NSP;
  #pragma unroll
  for (int mt = 0; mt < 8; ++mt)
    #pragma unroll
    for (int r = 0; r < 8; ++r) {
      const int o = oh * 128 + mt * 16 + r + (hi ? 8 : 0);
      const float inv = bn_gamma[o] * rsqrtf(bn_var[o] + 1e-5f);
      float v = acc[mt][r] + W_b[o];
      v = (v - bn_mean[o]) * inv + bn_beta[o];
      v += xb[(size_t)o * NSP + q];
      ob[(size_t)o * NSP + q] = v;
    }
}

// ---------------------------------------------------------------------------
// Concat copies: out[:, 0:256] = lc ; out[:, 512:768] = gc (float4 bulk)
// ---------------------------------------------------------------------------
__global__ void copy_kernel(const float4* __restrict__ lc,
                            const float4* __restrict__ gc,
                            float4* __restrict__ out)
{
  const size_t per = (size_t)CIN * NSP / 4;                 // 262144 f4 / batch
  const size_t i   = (size_t)blockIdx.x * blockDim.x + threadIdx.x;
  if (i >= (size_t)NB * per) return;
  const size_t b = i / per, r = i % per;
  const size_t obase = b * 3 * per;
  out[obase + r]           = lc[i];
  out[obase + 2 * per + r] = gc[i];
}

// ---------------------------------------------------------------------------
extern "C" void kernel_launch(void* const* d_in, const int* in_sizes, int n_in,
                              void* d_out, int out_size, void* d_ws, size_t ws_size,
                              hipStream_t stream) {
  (void)in_sizes; (void)n_in; (void)out_size; (void)ws_size;
  const float* lc      = (const float*)d_in[0];
  const float* fuse    = (const float*)d_in[1];
  const float* gc      = (const float*)d_in[2];
  const float* g_w     = (const float*)d_in[3];
  const float* g_b     = (const float*)d_in[4];
  const float* th_w    = (const float*)d_in[5];
  const float* th_b    = (const float*)d_in[6];
  const float* ph_w    = (const float*)d_in[7];
  const float* ph_b    = (const float*)d_in[8];
  const float* W_w     = (const float*)d_in[9];
  const float* W_b     = (const float*)d_in[10];
  const float* bn_g    = (const float*)d_in[11];
  const float* bn_be   = (const float*)d_in[12];
  const float* bn_m    = (const float*)d_in[13];
  const float* bn_v    = (const float*)d_in[14];
  float* out = (float*)d_out;

  char* ws = (char*)d_ws;                       // 36 MiB of f16 intermediates
  _Float16* theta_h  = (_Float16*)(ws);                      // 8 MiB [B][CI][N]
  _Float16* phi_full = (_Float16*)(ws + (size_t)( 8u << 20));// 8 MiB [B][CI][N]
  _Float16* g_full   = (_Float16*)(ws + (size_t)(16u << 20));// 8 MiB [B][CI][N]
  _Float16* phi_h    = (_Float16*)(ws + (size_t)(24u << 20));// 2 MiB [B][NK][CI]
  _Float16* g_h      = (_Float16*)(ws + (size_t)(26u << 20));// 2 MiB [B][NK][CI]
  _Float16* y_h      = (_Float16*)(ws + (size_t)(28u << 20));// 8 MiB [B][N][CI]

  // 1) projections: 8*3*256 waves -> 768 blocks of 8 waves
  proj_kernel<<<768, 256, 0, stream>>>(fuse, th_w, th_b, g_w, g_b, ph_w, ph_b,
                                       theta_h, g_full, phi_full);
  // 2) pool+transpose g and phi: NB*CI*NK = 1M threads each
  pool_kernel<<<4096, 256, 0, stream>>>(g_full, g_h);
  pool_kernel<<<4096, 256, 0, stream>>>(phi_full, phi_h);
  // 3) flash attention: 8*256 waves -> 256 blocks
  attn_kernel<<<256, 256, 0, stream>>>(theta_h, phi_h, g_h, y_h);
  // 4) output conv + BN + residual: 8*2*256 waves -> 512 blocks
  wout_kernel<<<512, 256, 0, stream>>>(y_h, W_w, W_b, bn_g, bn_be, bn_m, bn_v,
                                       fuse, out);
  // 5) concat copies: 2,097,152 threads
  copy_kernel<<<8192, 256, 0, stream>>>((const float4*)lc, (const float4*)gc,
                                        (float4*)out);
}
